// CausalSelfAttention_26379689132858
// MI455X (gfx1250) — matrix operand order repaired and checked by reading, not compiled, not run on hardware
//
#include <hip/hip_runtime.h>
#include <hip/hip_bf16.h>
#include <math.h>

// Problem constants (from reference): B=4, T=2048, E=768, H=12, D=64
#define Bsz 4
#define Tq  2048
#define Ed  768
#define Hh  12
#define Dd  64

typedef __attribute__((ext_vector_type(16))) _Float16 v16h;
typedef __attribute__((ext_vector_type(8)))  float    v8f;
typedef __attribute__((ext_vector_type(4)))  float    v4f;

union FragH { v16h v; v4f q[2]; _Float16 h[16]; };
union FragF { v8f  v; float f[8]; };

// ---------------------------------------------------------------------------
// CDNA5 async global->LDS copy (ASYNCcnt-tracked, CDNA5 ISA).
// VDST carries the per-lane LDS byte offset (low 32 bits of the generic LDS
// address), VADDR the 64-bit global address.
// ---------------------------------------------------------------------------
__device__ inline void async_ld_b128(_Float16* ldst, const _Float16* gsrc) {
  unsigned loff = (unsigned)(unsigned long long)(void*)ldst;
  asm volatile("global_load_async_to_lds_b128 %0, %1, off"
               :: "v"(loff), "v"(gsrc) : "memory");
}
#define WAIT_ASYNC(n) asm volatile("s_wait_asynccnt " #n ::: "memory")

// ---------------------------------------------------------------------------
// WMMA fragment loaders (CDNA5 16x16x32 f16, wave32; layouts per ISA 7.12.2)
// A (16x32): lane L<16 -> row M=L, holds K = {8*hi..8*hi+7, 16+8*hi..+7}
// B (32x16): lane L -> col N=L&15, holds K = {8*hi..8*hi+7, 16+8*hi..+7}
// Both load as two 16B vectors from row-major storage (B from B^T row-major).
// ---------------------------------------------------------------------------
__device__ inline v16h load_a16(const _Float16* tile, int ld) {
  const int lane = threadIdx.x & 31;
  const int r  = lane & 15;
  const int hi = lane >> 4;
  const _Float16* p = tile + r * ld + hi * 8;
  FragH u;
  u.q[0] = *(const v4f*)(p);
  u.q[1] = *(const v4f*)(p + 16);
  return u.v;
}

__device__ inline v16h load_b16(const _Float16* bt, int ld) {
  const int lane = threadIdx.x & 31;
  const int n  = lane & 15;
  const int hi = lane >> 4;
  const _Float16* p = bt + n * ld + hi * 8;
  FragH u;
  u.q[0] = *(const v4f*)(p);
  u.q[1] = *(const v4f*)(p + 16);
  return u.v;
}

__device__ inline v8f wmma_f16f32(v16h a, v16h b, v8f c) {
  return __builtin_amdgcn_wmma_f32_16x16x32_f16(
      /*neg_a=*/false, a, /*neg_b=*/false, b,
      /*c_mod=*/(short)0, c, /*reuse_a=*/false, /*reuse_b=*/false);
}

// ---------------------------------------------------------------------------
// Stage 1: fp32 -> f16 conversion
// ---------------------------------------------------------------------------
__global__ void cvt_f32_f16(const float* __restrict__ src,
                            _Float16* __restrict__ dst, int n) {
  int i = blockIdx.x * blockDim.x + threadIdx.x;
  if (i < n) dst[i] = (_Float16)src[i];
}

// ---------------------------------------------------------------------------
// Stage 2: QKV projection  Y = X * W^T + b  (nn.Linear with W = [out,in])
//   X: [M=B*T, 768] f16     W: [768,768] f16 (row o = output feature)
//   z = 0/1/2 -> Q/K/V.  Q,K stored [B,H,T,D]; V stored transposed [B,H,D,T]
// Block: 256 threads = 8 waves; wave w: rows m0 = bx*128 + 16w; cols by*64.
// Weight K-tiles (64 rows x 32 K) are shared by all 8 waves: staged in LDS
// via async global->LDS with double buffering; A fragments are register
// double-buffered so loads overlap WMMAs.
// ---------------------------------------------------------------------------
__global__ void __launch_bounds__(256)
qkv_gemm(const _Float16* __restrict__ Xh, const _Float16* __restrict__ Wh,
         const float* __restrict__ bq, const float* __restrict__ bk,
         const float* __restrict__ bv,
         _Float16* __restrict__ Qf, _Float16* __restrict__ Kf,
         _Float16* __restrict__ Vt)
{
  __shared__ __align__(16) _Float16 Wt[2][64 * 32];   // [buf][row_o][k]

  const int w    = threadIdx.x >> 5;
  const int lane = threadIdx.x & 31;
  const int col0 = lane & 15;
  const int hi   = lane >> 4;
  const int m0   = blockIdx.x * 128 + w * 16;
  const int n0   = blockIdx.y * 64;
  const int z    = blockIdx.z;

  const _Float16* W  = Wh + (size_t)z * Ed * Ed;
  const float* bias  = (z == 0) ? bq : (z == 1 ? bk : bv);

  const int trow = threadIdx.x >> 2;        // 0..63  (weight tile row)
  const int tc   = threadIdx.x & 3;         // 0..3   (16B chunk in row)

  FragF acc[4];
#pragma unroll
  for (int j = 0; j < 4; ++j)
#pragma unroll
    for (int p = 0; p < 8; ++p) acc[j].f[p] = 0.0f;

  // prologue: async-stage weight tile k=0, load first A fragment
  async_ld_b128(&Wt[0][trow * 32 + tc * 8],
                W + (size_t)(n0 + trow) * Ed + tc * 8);
  v16h a_cur = load_a16(Xh + (size_t)m0 * Ed, Ed);

  const int nk = Ed / 32;                   // 24 K-steps
  for (int ik = 0; ik < nk; ++ik) {
    const int k = ik * 32;
    __syncthreads();                        // prev iteration's LDS reads done
    if (ik + 1 < nk) {
      async_ld_b128(&Wt[(ik + 1) & 1][trow * 32 + tc * 8],
                    W + (size_t)(n0 + trow) * Ed + (k + 32) + tc * 8);
      WAIT_ASYNC(0x1);                      // oldest (tile ik) has landed
    } else {
      WAIT_ASYNC(0x0);
    }
    __syncthreads();                        // tile ik visible to all waves

    v16h a_nxt = a_cur;
    if (ik + 1 < nk)
      a_nxt = load_a16(Xh + (size_t)m0 * Ed + (k + 32), Ed);

    const _Float16* wt = &Wt[ik & 1][0];
#pragma unroll
    for (int j = 0; j < 4; ++j) {
      v16h bf = load_b16(wt + (16 * j) * 32, 32);
      acc[j].v = wmma_f16f32(a_cur, bf, acc[j].v);
    }
    a_cur = a_nxt;
  }

#pragma unroll
  for (int j = 0; j < 4; ++j) {
    const int o  = n0 + 16 * j + col0;       // output feature
    const float bval = bias[o];
    const int h = o >> 6;                    // head
    const int d = o & 63;                    // dim within head
#pragma unroll
    for (int p = 0; p < 8; ++p) {
      const int m = m0 + p + 8 * hi;         // global row = b*T + t
      const int b = m >> 11;
      const int t = m & (Tq - 1);
      const _Float16 val = (_Float16)(acc[j].f[p] + bval);
      const size_t bh = (size_t)(b * Hh + h);
      if (z == 0)      Qf[(bh * Tq + t) * Dd + d] = val;
      else if (z == 1) Kf[(bh * Tq + t) * Dd + d] = val;
      else             Vt[(bh * Dd + d) * Tq + t] = val;  // transposed
    }
  }
}

// ---------------------------------------------------------------------------
// Stage 3: causal flash attention.
// Grid: (T/128, B*H). Block: 256 threads = 8 waves; wave w owns 16 Q rows.
// Key blocks of 64; K tile [64 keys x 64 d] and V^T tile [64 d x 64 keys]
// staged in LDS by async global->LDS copies, double buffered: tile ib+1 is
// prefetched while tile ib is consumed (each wave issues 4 async ops/tile,
// s_wait_asynccnt 4 releases the previous tile — async loads complete
// in order). Online softmax; P routed through per-wave LDS staging.
// ---------------------------------------------------------------------------
__global__ void __launch_bounds__(256)
flash_attn(const _Float16* __restrict__ Qf, const _Float16* __restrict__ Kf,
           const _Float16* __restrict__ Vt, const float* __restrict__ amask,
           float* __restrict__ out)
{
  __shared__ __align__(16) _Float16 Ks[2][64 * 64];     // [buf][key][d]
  __shared__ __align__(16) _Float16 Vs[2][64 * 64];     // [buf][d][key]
  __shared__ __align__(16) _Float16 Ps[8][16 * 64];     // per-wave P staging

  const int w    = threadIdx.x >> 5;
  const int lane = threadIdx.x & 31;
  const int col  = lane & 15;
  const int hi   = lane >> 4;

  const int bh = blockIdx.y;            // 0..B*H-1
  const int b  = bh / Hh;
  const int h  = bh % Hh;
  const int q0 = blockIdx.x * 128;
  const int qr = q0 + w * 16;           // wave's first query row

  const _Float16* Qb = Qf + (size_t)bh * Tq * Dd;
  const _Float16* Kb = Kf + (size_t)bh * Tq * Dd;
  const _Float16* Vb = Vt + (size_t)bh * Dd * Tq;
  const float*    am = amask + (size_t)b * Tq;

  const v16h qa0 = load_a16(Qb + (size_t)qr * Dd + 0,  Dd);
  const v16h qa1 = load_a16(Qb + (size_t)qr * Dd + 32, Dd);

  FragF o_acc[4];
  float m_i[8], l_i[8];
#pragma unroll
  for (int j = 0; j < 4; ++j)
#pragma unroll
    for (int p = 0; p < 8; ++p) o_acc[j].f[p] = 0.0f;
#pragma unroll
  for (int p = 0; p < 8; ++p) { m_i[p] = -1e30f; l_i[p] = 0.0f; }

  const float sc = 0.125f;              // 1/sqrt(D), D=64
  const int nblk = (q0 + 128) / 64;     // causal: key blocks needed

  // Per-thread chunk indices for cooperative tile staging:
  //   K tile: 512 x 16B, contiguous from Kb + kb*64
  //   V tile: 512 x 16B, row d (stride T) from Vb + d*T + kb
  // Each thread covers chunks tid and tid+256 of each -> 4 async ops/wave.
  const int i0 = threadIdx.x;
  const int i1 = threadIdx.x + 256;

  // prologue: stage tile 0 into buffer 0
  {
    async_ld_b128(&Ks[0][i0 * 8], Kb + i0 * 8);
    async_ld_b128(&Ks[0][i1 * 8], Kb + i1 * 8);
    async_ld_b128(&Vs[0][i0 * 8], Vb + (size_t)(i0 >> 3) * Tq + (i0 & 7) * 8);
    async_ld_b128(&Vs[0][i1 * 8], Vb + (size_t)(i1 >> 3) * Tq + (i1 & 7) * 8);
  }

  for (int ib = 0; ib < nblk; ++ib) {
    const int kb  = ib * 64;
    const int cur = ib & 1;
    __syncthreads();                    // all waves done reading buf[cur^1]
    if (ib + 1 < nblk) {
      const int nb  = cur ^ 1;
      const int kb1 = kb + 64;
      async_ld_b128(&Ks[nb][i0 * 8], Kb + (size_t)kb1 * Dd + i0 * 8);
      async_ld_b128(&Ks[nb][i1 * 8], Kb + (size_t)kb1 * Dd + i1 * 8);
      async_ld_b128(&Vs[nb][i0 * 8],
                    Vb + (size_t)(i0 >> 3) * Tq + kb1 + (i0 & 7) * 8);
      async_ld_b128(&Vs[nb][i1 * 8],
                    Vb + (size_t)(i1 >> 3) * Tq + kb1 + (i1 & 7) * 8);
      WAIT_ASYNC(0x4);                  // tile ib (4 oldest ops) complete
    } else {
      WAIT_ASYNC(0x0);
    }
    __syncthreads();                    // tile ib visible to all waves

    const _Float16* Kt = &Ks[cur][0];
    const _Float16* Vts = &Vs[cur][0];

    // ---- S = Q K^T (4 tiles of 16 keys, 2 WMMA each) ----
    FragF s[4];
#pragma unroll
    for (int n = 0; n < 4; ++n) {
      v16h kb0 = load_b16(Kt + (16 * n) * Dd + 0,  Dd);
      v16h kb1f = load_b16(Kt + (16 * n) * Dd + 32, Dd);
      FragF cc;
#pragma unroll
      for (int p = 0; p < 8; ++p) cc.f[p] = 0.0f;
      cc.v = wmma_f16f32(qa0, kb0, cc.v);
      cc.v = wmma_f16f32(qa1, kb1f, cc.v);
      s[n] = cc;
    }

    // ---- scale + causal mask + attention_mask ----
#pragma unroll
    for (int n = 0; n < 4; ++n) {
      const int key = kb + 16 * n + col;
      const float amv = am[key];
#pragma unroll
      for (int p = 0; p < 8; ++p) {
        const int t = qr + p + 8 * hi;
        const float v = s[n].f[p] * sc + amv;
        s[n].f[p] = (key <= t) ? v : -1e30f;
      }
    }

    // ---- online softmax: row max across 4 tiles + 16-lane group ----
    float cur_m[8];
#pragma unroll
    for (int p = 0; p < 8; ++p) {
      float v = fmaxf(fmaxf(s[0].f[p], s[1].f[p]),
                      fmaxf(s[2].f[p], s[3].f[p]));
      v = fmaxf(v, __shfl_xor(v, 1, 32));
      v = fmaxf(v, __shfl_xor(v, 2, 32));
      v = fmaxf(v, __shfl_xor(v, 4, 32));
      v = fmaxf(v, __shfl_xor(v, 8, 32));
      cur_m[p] = v;
    }
    float corr[8];
#pragma unroll
    for (int p = 0; p < 8; ++p) {
      const float mn = fmaxf(m_i[p], cur_m[p]);
      corr[p] = __expf(m_i[p] - mn);
      m_i[p] = mn;
    }

    // ---- P = exp(S - m), row sums ----
    float rs[8];
#pragma unroll
    for (int p = 0; p < 8; ++p) rs[p] = 0.0f;
#pragma unroll
    for (int n = 0; n < 4; ++n)
#pragma unroll
      for (int p = 0; p < 8; ++p) {
        const float e = __expf(s[n].f[p] - m_i[p]);
        s[n].f[p] = e;
        rs[p] += e;
      }
#pragma unroll
    for (int p = 0; p < 8; ++p) {
      float v = rs[p];
      v += __shfl_xor(v, 1, 32);
      v += __shfl_xor(v, 2, 32);
      v += __shfl_xor(v, 4, 32);
      v += __shfl_xor(v, 8, 32);
      l_i[p] = l_i[p] * corr[p] + v;
    }
#pragma unroll
    for (int j = 0; j < 4; ++j)
#pragma unroll
      for (int p = 0; p < 8; ++p) o_acc[j].f[p] *= corr[p];

    // ---- C-layout P -> LDS -> A-fragment layout (same wave) ----
    _Float16* Pw = &Ps[w][0];
#pragma unroll
    for (int n = 0; n < 4; ++n)
#pragma unroll
      for (int p = 0; p < 8; ++p)
        Pw[(p + 8 * hi) * 64 + 16 * n + col] = (_Float16)s[n].f[p];

    // cross-lane LDS RAW within the wave: wait for stores explicitly
    asm volatile("s_wait_dscnt 0" ::: "memory");

    const v16h pa0 = load_a16(Pw + 0,  64);
    const v16h pa1 = load_a16(Pw + 32, 64);

    // ---- O += P V (4 output d-tiles, 2 WMMA each) ----
#pragma unroll
    for (int j = 0; j < 4; ++j) {
      v16h vb0 = load_b16(Vts + (16 * j) * 64 + 0,  64);
      v16h vb1 = load_b16(Vts + (16 * j) * 64 + 32, 64);
      o_acc[j].v = wmma_f16f32(pa0, vb0, o_acc[j].v);
      o_acc[j].v = wmma_f16f32(pa1, vb1, o_acc[j].v);
    }
  }

  // ---- epilogue: normalize and write fp32 output [B,T,H*D] ----
#pragma unroll
  for (int j = 0; j < 4; ++j) {
    const int d = 16 * j + col;
#pragma unroll
    for (int p = 0; p < 8; ++p) {
      const int t = qr + p + 8 * hi;
      out[((size_t)(b * Tq + t)) * Ed + h * Dd + d] = o_acc[j].f[p] / l_i[p];
    }
  }
}

// ---------------------------------------------------------------------------
// Host launcher
// ---------------------------------------------------------------------------
extern "C" void kernel_launch(void* const* d_in, const int* in_sizes, int n_in,
                              void* d_out, int out_size, void* d_ws,
                              size_t ws_size, hipStream_t stream) {
  const float* hidden = (const float*)d_in[0];
  const float* amask  = (const float*)d_in[1];
  const float* Wq     = (const float*)d_in[2];
  const float* bq     = (const float*)d_in[3];
  const float* Wk     = (const float*)d_in[4];
  const float* bk     = (const float*)d_in[5];
  const float* Wv     = (const float*)d_in[6];
  const float* bv     = (const float*)d_in[7];
  float* out = (float*)d_out;

  const size_t nX = (size_t)Bsz * Tq * Ed;   // 6,291,456
  const size_t nW = (size_t)Ed * Ed;         //   589,824
  const size_t nQ = nX;                      // B*H*T*D == B*T*E

  _Float16* ws = (_Float16*)d_ws;
  _Float16* Xh = ws;                 // hidden in f16
  _Float16* Wh = Xh + nX;            // [3][768*768] f16
  _Float16* Qf = Wh + 3 * nW;        // [B,H,T,D]
  _Float16* Kf = Qf + nQ;            // [B,H,T,D]
  _Float16* Vt = Kf + nQ;            // [B,H,D,T]

  const int thr = 256;
  cvt_f32_f16<<<(int)((nX + thr - 1) / thr), thr, 0, stream>>>(hidden, Xh, (int)nX);
  cvt_f32_f16<<<(int)((nW + thr - 1) / thr), thr, 0, stream>>>(Wq, Wh + 0 * nW, (int)nW);
  cvt_f32_f16<<<(int)((nW + thr - 1) / thr), thr, 0, stream>>>(Wk, Wh + 1 * nW, (int)nW);
  cvt_f32_f16<<<(int)((nW + thr - 1) / thr), thr, 0, stream>>>(Wv, Wh + 2 * nW, (int)nW);

  dim3 gGemm(Bsz * Tq / 128, Ed / 64, 3);    // (64, 12, 3)
  qkv_gemm<<<gGemm, 256, 0, stream>>>(Xh, Wh, bq, bk, bv, Qf, Kf, Vt);

  dim3 gAttn(Tq / 128, Bsz * Hh);            // (16, 48)
  flash_attn<<<gAttn, 256, 0, stream>>>(Qf, Kf, Vt, amask, out);
}
